// LuongAttention_12610023981417
// MI455X (gfx1250) — matrix operand order repaired and checked by reading, not compiled, not run on hardware
//
#include <hip/hip_runtime.h>
#include <hip/hip_bf16.h>

typedef __attribute__((ext_vector_type(16))) _Float16 v16h;
typedef __attribute__((ext_vector_type(8)))  _Float16 v8h;
typedef __attribute__((ext_vector_type(4)))  _Float16 v4h;
typedef __attribute__((ext_vector_type(8)))  float    v8f;

#define BATCH 8
#define TQ 2048
#define TK 2048
#define DIM 128

// ---------------- pre-pass: f32 -> f16 cast (vector x4) ----------------
__global__ void cvt_f32_f16_kernel(const float* __restrict__ src,
                                   _Float16* __restrict__ dst, int n4) {
  int i = blockIdx.x * blockDim.x + threadIdx.x;
  if (i < n4) {
    float4 f = ((const float4*)src)[i];
    v4h h;
    h[0] = (_Float16)f.x; h[1] = (_Float16)f.y;
    h[2] = (_Float16)f.z; h[3] = (_Float16)f.w;
    ((v4h*)dst)[i] = h;
  }
}

// ---------------- pre-pass: encT[b][d][k] = (f16)enc[b][k][d] ----------
// consecutive threads along k -> coalesced 2B writes; strided f32 reads hit L2.
__global__ void transpose_enc_kernel(const float* __restrict__ enc,
                                     _Float16* __restrict__ encT) {
  int k = blockIdx.x * blockDim.x + threadIdx.x; // fastest
  int d = blockIdx.y;
  int b = blockIdx.z;
  encT[((size_t)b * DIM + d) * TK + k] =
      (_Float16)enc[((size_t)b * TK + k) * DIM + d];
}

// ---------------- fused (dec @ encT) @ enc, streaming over k -----------
// One wave: 16 q-rows x 128 d-cols of output.
// GEMM1 computes S^T tiles so its D-register layout IS GEMM2's A layout.
__launch_bounds__(256)
__global__ void luong_fused_kernel(const _Float16* __restrict__ dec16,
                                   const _Float16* __restrict__ enc16,
                                   const _Float16* __restrict__ encT,
                                   float* __restrict__ out) {
  const int lane = threadIdx.x & 31;
  const int wave = threadIdx.x >> 5;
  const int n = lane & 15;  // N index (col within 16-wide tile)
  const int h = lane >> 4;  // half-wave select

  const int b    = blockIdx.x >> 4;       // 16 q-blocks of 128 rows per batch
  const int qblk = blockIdx.x & 15;
  const int qbase = qblk * 128 + wave * 16;

  const _Float16* decRow  = dec16 + ((size_t)(b * TQ + qbase + n)) * DIM;
  const _Float16* encBase = enc16 + (size_t)b * TK * DIM;
  const _Float16* encTB   = encT + (size_t)b * DIM * TK;

  // GEMM1 B operand (dec rows), loop invariant over k.
  // B layout: lane n = q-row m, element e -> reduction index d = c*32 + 16*h + e
  v16h b1[4];
#pragma unroll
  for (int c = 0; c < 4; ++c)
    b1[c] = *(const v16h*)(decRow + c * 32 + 16 * h);

  v8f acc[8];
#pragma unroll
  for (int dt = 0; dt < 8; ++dt) acc[dt] = (v8f){0.f,0.f,0.f,0.f,0.f,0.f,0.f,0.f};

  for (int kc = 0; kc < TK / 32; ++kc) {
    const int kbase = kc * 32;
    const _Float16* a1r0 = encBase + (size_t)(kbase + n) * DIM;       // S^T tile0 rows
    const _Float16* a1r1 = encBase + (size_t)(kbase + 16 + n) * DIM;  // S^T tile1 rows

    // prefetch next k-chunk's enc rows (speculative; dropped if invalid)
    __builtin_prefetch(a1r0 + 32 * DIM, 0, 1);
    __builtin_prefetch(a1r1 + 32 * DIM, 0, 1);

    v8f s0 = (v8f){0.f,0.f,0.f,0.f,0.f,0.f,0.f,0.f};
    v8f s1 = s0;
#pragma unroll
    for (int c = 0; c < 4; ++c) {
      // A layout (16-bit, 16x32): e -> d = c*32 + 16h + e
      v8h lo0 = *(const v8h*)(a1r0 + c * 32 + 16 * h);
      v8h hi0 = *(const v8h*)(a1r0 + c * 32 + 16 * h + 8);
      v8h lo1 = *(const v8h*)(a1r1 + c * 32 + 16 * h);
      v8h hi1 = *(const v8h*)(a1r1 + c * 32 + 16 * h + 8);
      v16h a0 = __builtin_shufflevector(lo0, hi0, 0,1,2,3,4,5,6,7,8,9,10,11,12,13,14,15);
      v16h a1 = __builtin_shufflevector(lo1, hi1, 0,1,2,3,4,5,6,7,8,9,10,11,12,13,14,15);
      s0 = __builtin_amdgcn_wmma_f32_16x16x32_f16(false, a0, false, b1[c],
                                                  (short)0, s0, false, false);
      s1 = __builtin_amdgcn_wmma_f32_16x16x32_f16(false, a1, false, b1[c],
                                                  (short)0, s1, false, false);
    }

    // s0/s1 D-layout == GEMM2 A-operand layout for this 32-wide k chunk:
    // lane(n=m,h): s0[r] = S[m][kbase+8h+r], s1[r] = S[m][kbase+16+8h+r]
    v16h a2;
#pragma unroll
    for (int e = 0; e < 8; ++e) {
      a2[e]     = (_Float16)s0[e];
      a2[8 + e] = (_Float16)s1[e];
    }

    // GEMM2: acc[dt] += S(16x32) x enc(32x16). B: lane n = d col, e<8 -> k = kbase + 8h + e ; e>=8 -> k = kbase + 16 + 8h + (e-8)
#pragma unroll
    for (int dt = 0; dt < 8; ++dt) {
      const _Float16* bp = encTB + (size_t)(dt * 16 + n) * TK + kbase;
      v8h blo = *(const v8h*)(bp + 8 * h);
      v8h bhi = *(const v8h*)(bp + 16 + 8 * h);
      v16h b2 = __builtin_shufflevector(blo, bhi, 0,1,2,3,4,5,6,7,8,9,10,11,12,13,14,15);
      acc[dt] = __builtin_amdgcn_wmma_f32_16x16x32_f16(false, a2, false, b2,
                                                       (short)0, acc[dt], false, false);
    }
  }

  // D layout: VGPR r -> row m = r + 8h, col d = dt*16 + n
  float* outBase = out + ((size_t)(b * TQ + qbase)) * DIM;
#pragma unroll
  for (int dt = 0; dt < 8; ++dt)
#pragma unroll
    for (int r = 0; r < 8; ++r)
      outBase[(size_t)(r + 8 * h) * DIM + dt * 16 + n] = acc[dt][r];
}

extern "C" void kernel_launch(void* const* d_in, const int* in_sizes, int n_in,
                              void* d_out, int out_size, void* d_ws, size_t ws_size,
                              hipStream_t stream) {
  (void)in_sizes; (void)n_in; (void)out_size; (void)ws_size;
  const float* enc = (const float*)d_in[0];  // encoder_hidden_states (B,TK,D)
  const float* dec = (const float*)d_in[1];  // decoder_hidden_states (B,TQ,D)
  float* out = (float*)d_out;

  const size_t nElem = (size_t)BATCH * TQ * DIM;  // 2,097,152 per tensor
  _Float16* dec16 = (_Float16*)d_ws;
  _Float16* enc16 = dec16 + nElem;
  _Float16* encT  = enc16 + nElem;  // total ws: 3 * 4 MB = 12 MB

  const int n4 = (int)(nElem / 4);
  cvt_f32_f16_kernel<<<(n4 + 255) / 256, 256, 0, stream>>>(dec, dec16, n4);
  cvt_f32_f16_kernel<<<(n4 + 255) / 256, 256, 0, stream>>>(enc, enc16, n4);
  transpose_enc_kernel<<<dim3(TK / 256, DIM, BATCH), 256, 0, stream>>>(enc, encT);

  luong_fused_kernel<<<BATCH * (TQ / 128), 256, 0, stream>>>(dec16, enc16, encT, out);
}
